// LaneATTHead_28235115004490
// MI455X (gfx1250) — matrix operand order repaired and checked by reading, not compiled, not run on hardware
//
#include <hip/hip_runtime.h>
#include <hip/hip_bf16.h>

// ---------------- problem constants ----------------
#define BB     8
#define CIN    512
#define CF     64
#define HF     10
#define WF     25
#define HWF    (HF * WF)        // 250
#define NA     2784             // anchors
#define NM     2783             // N-1 (attention targets)
#define DD     640              // CF*HF
#define D2     1280
#define LANEK  77               // 5 + 72 per lane output row

typedef __attribute__((ext_vector_type(16))) __bf16 v16bf;
typedef __attribute__((ext_vector_type(8)))  float  v8f;

union Frag {
    uint4          q[2];
    v16bf          v;
    unsigned short s[16];
};

__device__ __forceinline__ unsigned short f32_to_bf16(float f) {
    union { float f; unsigned u; } c; c.f = f;
    unsigned r = c.u + 0x7FFFu + ((c.u >> 16) & 1u);   // round-to-nearest-even
    return (unsigned short)(r >> 16);
}
__device__ __forceinline__ float bf16_to_f32(unsigned short h) {
    union { unsigned u; float f; } c; c.u = ((unsigned)h) << 16;
    return c.f;
}

// ---------------- 1) 1x1 conv: feat[b,f,p] = x[b,:,p] . conv_w[f,:] + conv_b[f] ----
__global__ void conv1x1_kernel(const float* __restrict__ x,
                               const float* __restrict__ w,
                               const float* __restrict__ bias,
                               float* __restrict__ feat) {
    int gid = blockIdx.x * blockDim.x + threadIdx.x;     // 128000 total
    int p = gid % HWF;
    int f = (gid / HWF) % CF;
    int b = gid / (HWF * CF);
    const float* xp = x + (size_t)b * CIN * HWF + p;
    const float* wp = w + (size_t)f * CIN;
    float s = bias[f];
#pragma unroll 8
    for (int c = 0; c < CIN; ++c)
        s += xp[(size_t)c * HWF] * wp[c];
    feat[gid] = s;
}

// ---------------- 2) convert attn_w to bf16 ----------------
__global__ void cvt_attw_kernel(const float* __restrict__ aw,
                                unsigned short* __restrict__ awb, int n) {
    int gid = blockIdx.x * blockDim.x + threadIdx.x;
    if (gid < n) awb[gid] = f32_to_bf16(aw[gid]);
}

// ---------------- 3) gather pf (bf16, row-major + transposed) ----------------
// pf[b,n,d] with d = c*HF + y ; element = invalid[n,y] ? 0 : feat[b,c,y,cut_xs[n,y]]
__global__ void gather_kernel(const float* __restrict__ feat,
                              const int* __restrict__ cut_xs,
                              const int* __restrict__ invalid,
                              unsigned short* __restrict__ pf_bf,
                              unsigned short* __restrict__ pfT_bf) {
    int gid = blockIdx.x * blockDim.x + threadIdx.x;     // BB*NA*DD total
    int d = gid % DD;
    int n = (gid / DD) % NA;
    int b = gid / (DD * NA);
    int c = d / HF, y = d % HF;
    int xs  = cut_xs[n * HF + y];
    int inv = invalid[n * HF + y];
    float v = inv ? 0.0f : feat[(((size_t)b * CF + c) * HF + y) * WF + xs];
    unsigned short h = f32_to_bf16(v);
    pf_bf[gid] = h;
    pfT_bf[((size_t)b * DD + d) * NA + n] = h;
}

// ---------------- 4) scores GEMM (bf16 WMMA, f32 acc) ----------------
// scores[b,n,m] = pf[b,n,:] . attn_w[m,:] + attn_b[m], written scattered into
// attn[b,n, m + (m>=n)] (the attn_mat output region). Diagonal left for softmax.
__global__ void scores_wmma_kernel(const unsigned short* __restrict__ pf_bf,
                                   const unsigned short* __restrict__ attw_bf,
                                   const float* __restrict__ attn_b,
                                   float* __restrict__ attn) {
    const int lane = threadIdx.x & 31;
    const int wave = threadIdx.x >> 5;
    const int half = lane >> 4;
    const int lcol = lane & 15;
    const int b  = blockIdx.z;
    const int n0 = blockIdx.y * 16;
    const int m0 = blockIdx.x * 128 + wave * 16;
    if (m0 >= NM) return;                               // no barriers in kernel

    const int mB = (m0 + lcol < NM) ? (m0 + lcol) : (NM - 1);
    // A: 16x32 bf16, lane holds row n0+lcol, K runs {s..s+7} and {s+16..s+23}, s = half*8
    const unsigned short* ap = pf_bf + ((size_t)(b * NA + n0 + lcol)) * DD + half * 8;
    // B: 32x16 bf16, lane holds column m, K runs {s..s+7} and {s+16..s+23}, s = half*8
    const unsigned short* bp = attw_bf + (size_t)mB * DD + half * 8;

    v8f acc = {0.f, 0.f, 0.f, 0.f, 0.f, 0.f, 0.f, 0.f};
    for (int kk = 0; kk < DD; kk += 32) {
        __builtin_prefetch(ap + kk + 64, 0, 1);
        __builtin_prefetch(bp + kk + 64, 0, 1);
        Frag a, bf;
        a.q[0]  = *(const uint4*)(ap + kk);
        a.q[1]  = *(const uint4*)(ap + kk + 16);
        bf.q[0] = *(const uint4*)(bp + kk);
        bf.q[1] = *(const uint4*)(bp + kk + 16);
        acc = __builtin_amdgcn_wmma_f32_16x16x32_bf16(
                  false, a.v, false, bf.v, (short)0, acc, false, false);
    }
    const int m = m0 + lcol;
    if (m < NM) {
        const float bias = attn_b[m];
#pragma unroll
        for (int r = 0; r < 8; ++r) {
            int ni  = n0 + r + half * 8;                // C/D layout: M = r + half*8
            int col = m + (m >= ni);
            attn[((size_t)(b * NA + ni)) * NA + col] = acc[r] + bias;
        }
    }
}

// ---------------- 5) softmax in-place over each attn row (diag -> 0) ----------------
__global__ void softmax_kernel(float* __restrict__ attn) {
    const int row = blockIdx.x;                         // b*NA + n
    const int n   = row % NA;
    float* s = attn + (size_t)row * NA;
    __shared__ float red[256];
    const int tid = threadIdx.x;

    float v[11];
    float mx = -INFINITY;
#pragma unroll
    for (int i = 0; i < 11; ++i) {
        int j = tid + i * 256;
        float t = -INFINITY;
        if (j < NA && j != n) t = s[j];
        v[i] = t;
        mx = fmaxf(mx, t);
    }
    red[tid] = mx; __syncthreads();
    for (int o = 128; o > 0; o >>= 1) {
        if (tid < o) red[tid] = fmaxf(red[tid], red[tid + o]);
        __syncthreads();
    }
    mx = red[0]; __syncthreads();

    float sum = 0.f;
#pragma unroll
    for (int i = 0; i < 11; ++i) {
        float e = (v[i] == -INFINITY) ? 0.f : __expf(v[i] - mx);
        v[i] = e;
        sum += e;
    }
    red[tid] = sum; __syncthreads();
    for (int o = 128; o > 0; o >>= 1) {
        if (tid < o) red[tid] += red[tid + o];
        __syncthreads();
    }
    const float inv = 1.0f / red[0];
#pragma unroll
    for (int i = 0; i < 11; ++i) {
        int j = tid + i * 256;
        if (j < NA) s[j] = v[i] * inv;                  // j==n writes 0
    }
}

// ---------------- 6) att_feats GEMM: attn (f32 -> bf16) @ pfT (bf16) ----------------
__global__ void attfeat_wmma_kernel(const float* __restrict__ attn,
                                    const unsigned short* __restrict__ pfT_bf,
                                    float* __restrict__ attf) {
    const int lane = threadIdx.x & 31;
    const int wave = threadIdx.x >> 5;
    const int half = lane >> 4;
    const int lcol = lane & 15;
    const int b  = blockIdx.z;
    const int n0 = blockIdx.y * 16;
    const int d0 = blockIdx.x * 128 + wave * 16;        // 5*128 = 640 exact

    const float* ap = attn + ((size_t)(b * NA + n0 + lcol)) * NA + half * 8;
    const unsigned short* bp = pfT_bf + ((size_t)(b * DD + d0 + lcol)) * NA + half * 8;

    v8f acc = {0.f, 0.f, 0.f, 0.f, 0.f, 0.f, 0.f, 0.f};
    for (int kk = 0; kk < NA; kk += 32) {
        __builtin_prefetch(ap + kk + 64, 0, 1);
        __builtin_prefetch(bp + kk + 64, 0, 1);
        float4 f0 = *(const float4*)(ap + kk);
        float4 f1 = *(const float4*)(ap + kk + 4);
        float4 f2 = *(const float4*)(ap + kk + 16);
        float4 f3 = *(const float4*)(ap + kk + 20);
        Frag a, bf;
        a.s[0]  = f32_to_bf16(f0.x); a.s[1]  = f32_to_bf16(f0.y);
        a.s[2]  = f32_to_bf16(f0.z); a.s[3]  = f32_to_bf16(f0.w);
        a.s[4]  = f32_to_bf16(f1.x); a.s[5]  = f32_to_bf16(f1.y);
        a.s[6]  = f32_to_bf16(f1.z); a.s[7]  = f32_to_bf16(f1.w);
        a.s[8]  = f32_to_bf16(f2.x); a.s[9]  = f32_to_bf16(f2.y);
        a.s[10] = f32_to_bf16(f2.z); a.s[11] = f32_to_bf16(f2.w);
        a.s[12] = f32_to_bf16(f3.x); a.s[13] = f32_to_bf16(f3.y);
        a.s[14] = f32_to_bf16(f3.z); a.s[15] = f32_to_bf16(f3.w);
        bf.q[0] = *(const uint4*)(bp + kk);
        bf.q[1] = *(const uint4*)(bp + kk + 16);
        acc = __builtin_amdgcn_wmma_f32_16x16x32_bf16(
                  false, a.v, false, bf.v, (short)0, acc, false, false);
    }
#pragma unroll
    for (int r = 0; r < 8; ++r) {
        int ni = n0 + r + half * 8;
        attf[((size_t)(b * NA + ni)) * DD + d0 + lcol] = acc[r];
    }
}

// ---------------- 7) heads: cls(2) + reg(73) over cat=[att_feats, pf], lane assembly --
__global__ void heads_kernel(const float* __restrict__ attf,
                             const unsigned short* __restrict__ pf_bf,
                             const float* __restrict__ cls_w,
                             const float* __restrict__ cls_b,
                             const float* __restrict__ reg_w,
                             const float* __restrict__ reg_b,
                             const float* __restrict__ anchors,
                             float* __restrict__ out_cls,
                             float* __restrict__ out_lanes) {
    __shared__ float catS[8][D2];
    __shared__ float red[8][80];
    const int wave = threadIdx.x >> 5;
    const int lane = threadIdx.x & 31;
    const int idx  = blockIdx.x * 8 + wave;             // < BB*NA (grid exact)
    const int n    = idx % NA;

    const float* af = attf + (size_t)idx * DD;
    const unsigned short* pr = pf_bf + (size_t)idx * DD;
    for (int d = lane; d < DD; d += 32) {
        catS[wave][d]      = af[d];
        catS[wave][DD + d] = bf16_to_f32(pr[d]);
    }
    __syncthreads();

    for (int c = 0; c < 75; ++c) {
        const float* w = (c < 2) ? (cls_w + (size_t)c * D2)
                                 : (reg_w + (size_t)(c - 2) * D2);
        float p = 0.f;
        for (int d = lane; d < D2; d += 32) p += catS[wave][d] * w[d];
        for (int off = 16; off > 0; off >>= 1) p += __shfl_down(p, off, 32);
        if (lane == 0)
            red[wave][c] = p + ((c < 2) ? cls_b[c] : reg_b[c - 2]);
    }
    __syncthreads();

    if (lane < 2) out_cls[(size_t)idx * 2 + lane] = red[wave][lane];
    for (int k = lane; k < LANEK; k += 32) {
        float a = anchors[(size_t)n * LANEK + k];
        float v;
        if (k < 4)       v = a;
        else if (k == 4) v = red[wave][2];               // reg[...,0]
        else             v = a + red[wave][2 + (k - 4)]; // anchors[5:] + reg[...,1:]
        out_lanes[(size_t)idx * LANEK + k] = v;
    }
}

// ---------------- launch ----------------
extern "C" void kernel_launch(void* const* d_in, const int* in_sizes, int n_in,
                              void* d_out, int out_size, void* d_ws, size_t ws_size,
                              hipStream_t stream) {
    const float* x       = (const float*)d_in[0];
    const float* conv_w  = (const float*)d_in[1];
    const float* conv_b  = (const float*)d_in[2];
    const float* attn_w  = (const float*)d_in[3];
    const float* attn_b  = (const float*)d_in[4];
    const float* cls_w   = (const float*)d_in[5];
    const float* cls_b   = (const float*)d_in[6];
    const float* reg_w   = (const float*)d_in[7];
    const float* reg_b   = (const float*)d_in[8];
    const float* anchors = (const float*)d_in[9];
    const int*   cut_xs  = (const int*)d_in[10];
    const int*   invalid = (const int*)d_in[11];
    float* out = (float*)d_out;

    // workspace carve-up (all 256B-aligned by construction)
    char* ws = (char*)d_ws;
    const size_t featB  = (size_t)BB * CF * HWF * 4;         //    512,000
    const size_t pfB    = (size_t)BB * NA * DD * 2;          // 28,508,160
    const size_t attwB  = (size_t)NM * DD * 2;               //  3,562,240
    float*          feat    = (float*)ws;
    unsigned short* pf_bf   = (unsigned short*)(ws + featB);
    unsigned short* pfT_bf  = (unsigned short*)(ws + featB + pfB);
    unsigned short* attw_bf = (unsigned short*)(ws + featB + 2 * pfB);
    float*          attf    = (float*)(ws + featB + 2 * pfB + attwB);

    // output regions: cls | lanes | attn_mat
    float* out_cls   = out;
    float* out_lanes = out + (size_t)BB * NA * 2;
    float* attn      = out + (size_t)BB * NA * 2 + (size_t)BB * NA * LANEK;

    conv1x1_kernel<<<(BB * CF * HWF) / 256, 256, 0, stream>>>(x, conv_w, conv_b, feat);

    const int awN = NM * DD;
    cvt_attw_kernel<<<(awN + 255) / 256, 256, 0, stream>>>(attn_w, attw_bf, awN);

    gather_kernel<<<(BB * NA * DD) / 256, 256, 0, stream>>>(feat, cut_xs, invalid,
                                                            pf_bf, pfT_bf);

    scores_wmma_kernel<<<dim3(22, NA / 16, BB), 256, 0, stream>>>(pf_bf, attw_bf,
                                                                  attn_b, attn);

    softmax_kernel<<<BB * NA, 256, 0, stream>>>(attn);

    attfeat_wmma_kernel<<<dim3(5, NA / 16, BB), 256, 0, stream>>>(attn, pfT_bf, attf);

    heads_kernel<<<(BB * NA) / 8, 256, 0, stream>>>(attf, pf_bf, cls_w, cls_b,
                                                    reg_w, reg_b, anchors,
                                                    out_cls, out_lanes);
}
